// ProxySR_17051020165526
// MI455X (gfx1250) — matrix operand order repaired and checked by reading, not compiled, not run on hardware
//
#include <hip/hip_runtime.h>
#include <hip/hip_bf16.h>

typedef __attribute__((ext_vector_type(16))) _Float16 v16h;
typedef __attribute__((ext_vector_type(8)))  float    v8f;

#define BB   32
#define TT   50
#define NN   50000
#define NROW (NN + 1)
#define DD   128
#define KK   16
#define MPP  50
#define HIDN 72
#define TPAD 64

static constexpr size_t DIST_ELEMS = (size_t)BB * TT * NROW;   // 80,001,600

// ---------------- workspace layout (bytes) ----------------
static constexpr size_t OFF_IF  = 0;                       // f16 I_emb   50001*128*2 = 12,800,256
static constexpr size_t OFF_IN2 = 12800256;                // float[50001]
static constexpr size_t OFF_XS  = 13000448;                // float[32*50*128]
static constexpr size_t OFF_Q   = OFF_XS  + 819200;
static constexpr size_t OFF_K   = OFF_Q   + 819200;
static constexpr size_t OFF_V   = OFF_K   + 819200;
static constexpr size_t OFF_H   = OFF_V   + 819200;
static constexpr size_t OFF_PS  = OFF_H   + 819200;        // float[32*128]
static constexpr size_t OFF_VV  = OFF_PS  + 16384;         // float[32*128]
static constexpr size_t OFF_A   = OFF_VV  + 16384;         // f16 Fp [32*64*128], row 50 = v_b
static constexpr size_t OFF_FN2 = OFF_A   + 524288;        // float[32*64]

__device__ __forceinline__ float block_reduce_sum(float v, float* red) {
    int tid = threadIdx.x;
    red[tid] = v; __syncthreads();
    for (int s = 64; s > 0; s >>= 1) { if (tid < s) red[tid] += red[tid + s]; __syncthreads(); }
    float r = red[0]; __syncthreads();
    return r;
}

// ---------------- 1) I_emb -> f16 + ||I_n||^2 ----------------
__global__ void k_prep(const float* __restrict__ I, _Float16* __restrict__ If, float* __restrict__ In2) {
    __shared__ float red[DD];
    int n = blockIdx.x, k = threadIdx.x;
    float v = I[(size_t)n * DD + k];
    If[(size_t)n * DD + k] = (_Float16)v;
    float s = block_reduce_sum(v * v, red);
    if (k == 0) In2[n] = s;
}

// ---------------- 2) ProxySelection ----------------
__global__ void k_proxy(const int* __restrict__ session, const int* __restrict__ lengths,
                        const int* __restrict__ tau,
                        const float* __restrict__ I, const float* __restrict__ P,
                        const float* __restrict__ EP, const float* __restrict__ V,
                        const float* __restrict__ WP1, const float* __restrict__ WP2,
                        float* __restrict__ ps_out, float* __restrict__ v_out,
                        float* __restrict__ orth_out) {
    int b = blockIdx.x, tid = threadIdx.x;
    __shared__ float xb[DD], hb[HIDN], red[DD], pis[KK], vsc[KK];
    float facc = 0.f;
    for (int t = 0; t < TT; ++t) {
        int sid = session[b * TT + t];
        float valid = (sid != 0) ? 1.f : 0.f;
        xb[tid] = valid * (I[(size_t)sid * DD + tid] + EP[(t + MPP - TT) * DD + tid]);
        __syncthreads();
        if (tid < HIDN) {
            float a = 0.f;
            for (int k = 0; k < DD; ++k) a += WP1[tid * DD + k] * xb[k];
            hb[tid] = (a >= 0.f) ? a : 0.1f * a;
        }
        __syncthreads();
        if (tid < KK) {
            float a = 0.f;
            for (int h = 0; h < HIDN; ++h) a += WP2[tid * HIDN + h] * hb[h];
            facc += a;
        }
        __syncthreads();
    }
    float lf = (float)lengths[b], tv = (float)tau[0];
    if (tid < KK) pis[tid] = facc / lf / tv;
    __syncthreads();
    if (tid == 0) {
        float mx = pis[0];
        for (int i = 1; i < KK; ++i) mx = fmaxf(mx, pis[i]);
        float sm = 0.f;
        for (int i = 0; i < KK; ++i) { pis[i] = __expf(pis[i] - mx); sm += pis[i]; }
        float inv = 1.f / sm;
        for (int i = 0; i < KK; ++i) pis[i] *= inv;
    }
    __syncthreads();
    if (tid < KK) {
        float s = 0.f;
        for (int k = 0; k < DD; ++k) { float x = V[tid * DD + k]; s += x * x; }
        vsc[tid] = 1.f / fmaxf(sqrtf(s), 1e-12f);
    }
    __syncthreads();
    float ps = 0.f, vr = 0.f;
    for (int k = 0; k < KK; ++k) {
        ps += pis[k] * P[k * DD + tid];
        vr += pis[k] * V[k * DD + tid] * vsc[k];
    }
    float vn2 = block_reduce_sum(vr * vr, red);
    float vv  = vr * (1.f / fmaxf(sqrtf(vn2), 1e-12f));
    v_out[b * DD + tid]  = vv;
    ps_out[b * DD + tid] = ps;
    float dvp = block_reduce_sum(vv * ps, red);
    float pn2 = block_reduce_sum(ps * ps, red);
    if (tid == 0) orth_out[b] = fabsf(dvp) / sqrtf(pn2);
}

// ---------------- 3) Xs, Q, K, V ----------------
__global__ void k_qkv(const int* __restrict__ session, const int* __restrict__ lengths,
                      const float* __restrict__ I, const float* __restrict__ ES,
                      const float* __restrict__ Wq, const float* __restrict__ Wk,
                      const float* __restrict__ Wv,
                      float* __restrict__ Xs, float* __restrict__ Q,
                      float* __restrict__ Kb, float* __restrict__ Vb) {
    int id = blockIdx.x, b = id / TT, t = id % TT, d = threadIdx.x;
    __shared__ float xb[DD];
    int sid = session[b * TT + t];
    int ei  = t + 1 - (TT - lengths[b]);
    ei = ei < 0 ? 0 : (ei > MPP ? MPP : ei);
    float x = I[(size_t)sid * DD + d] + ES[ei * DD + d];
    xb[d] = x; __syncthreads();
    float q = 0.f, k = 0.f, v = 0.f;
    for (int e = 0; e < DD; ++e) {
        float xe = xb[e];
        q += Wq[d * DD + e] * xe; k += Wk[d * DD + e] * xe; v += Wv[d * DD + e] * xe;
    }
    int o = (b * TT + t) * DD + d;
    Xs[o] = x; Q[o] = q; Kb[o] = k; Vb[o] = v;
}

// ---------------- 4) attention + Wo + LN1 ----------------
__global__ void k_attn(const int* __restrict__ session,
                       const float* __restrict__ Xs, const float* __restrict__ Q,
                       const float* __restrict__ Kb, const float* __restrict__ Vb,
                       const float* __restrict__ Wo,
                       const float* __restrict__ g1, const float* __restrict__ b1,
                       float* __restrict__ H) {
    int id = blockIdx.x, b = id / TT, t = id % TT, tid = threadIdx.x;
    __shared__ float qb[DD], sc[TT], cb[DD], red[DD];
    qb[tid] = Q[(b * TT + t) * DD + tid];
    __syncthreads();
    if (tid < TT) {
        int s = tid;
        bool masked = ((s > t) || (session[b * TT + s] == 0)) && (s != 0);
        float a = -1e9f;
        if (!masked) {
            a = 0.f;
            for (int e = 0; e < DD; ++e) a += qb[e] * Kb[(b * TT + s) * DD + e];
            a *= 0.08838834764831845f;   // 1/sqrt(128)
        }
        sc[s] = a;
    }
    __syncthreads();
    if (tid == 0) {
        float mx = sc[0];
        for (int i = 1; i < TT; ++i) mx = fmaxf(mx, sc[i]);
        float sm = 0.f;
        for (int i = 0; i < TT; ++i) { sc[i] = __expf(sc[i] - mx); sm += sc[i]; }
        float inv = 1.f / sm;
        for (int i = 0; i < TT; ++i) sc[i] *= inv;
    }
    __syncthreads();
    float c = 0.f;
    for (int s = 0; s < TT; ++s) c += sc[s] * Vb[(b * TT + s) * DD + tid];
    cb[tid] = c; __syncthreads();
    float ho = 0.f;
    for (int e = 0; e < DD; ++e) ho += Wo[tid * DD + e] * cb[e];
    float r  = ho + Xs[(b * TT + t) * DD + tid];
    float s1 = block_reduce_sum(r, red);
    float s2 = block_reduce_sum(r * r, red);
    float mean = s1 * (1.f / DD);
    float var  = s2 * (1.f / DD) - mean * mean;
    float hn   = (r - mean) * rsqrtf(var + 1e-6f) * g1[tid] + b1[tid];
    float np   = (session[b * TT + t] != 0) ? 1.f : 0.f;
    H[(b * TT + t) * DD + tid] = hn * np;
}

// ---------------- 5) FFN + LN2 + Fp/Fn2 (f16 A-matrix, padded to 64 rows) ----------------
// Row layout per batch b:  rows 0..49 = Fp[b,t,:],  row 50 = v_b  (so the WMMA GEMM
// produces vI = v_b . I_n as tile m=3 / tile-row 2 for free),  rows 51..63 = 0.
__global__ void k_ffn(const int* __restrict__ session, const float* __restrict__ H,
                      const float* __restrict__ F1, const float* __restrict__ F2,
                      const float* __restrict__ g2, const float* __restrict__ b2,
                      const float* __restrict__ psw, const float* __restrict__ vw,
                      _Float16* __restrict__ A, float* __restrict__ Fn2) {
    int id = blockIdx.x, b = id / TPAD, t = id % TPAD, tid = threadIdx.x;
    if (t >= TT) {
        float av = (t == TT) ? vw[b * DD + tid] : 0.f;
        A[((size_t)b * TPAD + t) * DD + tid] = (_Float16)av;
        if (tid == 0) Fn2[b * TPAD + t] = 0.f;
        return;
    }
    __shared__ float hb[DD], fb[DD], red[DD];
    float h = H[(b * TT + t) * DD + tid];
    hb[tid] = h; __syncthreads();
    float f1 = 0.f;
    for (int e = 0; e < DD; ++e) f1 += F1[tid * DD + e] * hb[e];
    fb[tid] = fmaxf(f1, 0.f); __syncthreads();
    float f2 = 0.f;
    for (int e = 0; e < DD; ++e) f2 += F2[tid * DD + e] * fb[e];
    float r  = f2 + h;
    float s1 = block_reduce_sum(r, red);
    float s2 = block_reduce_sum(r * r, red);
    float mean = s1 * (1.f / DD);
    float var  = s2 * (1.f / DD) - mean * mean;
    float np   = (session[b * TT + t] != 0) ? 1.f : 0.f;
    float ss   = ((r - mean) * rsqrtf(var + 1e-6f) * g2[tid] + b2[tid]) * np;
    float v  = vw[b * DD + tid];
    float ps = psw[b * DD + tid];
    float dvs = block_reduce_sum(v * ss, red);      // v . s_s
    float F   = ps + ss - v * dvs;                  // p_s + s_s_proj
    float Fv  = block_reduce_sum(v * F, red);       // F . v
    float Fp  = F - Fv * v;
    float fn2 = block_reduce_sum(F * F, red);
    A[((size_t)b * TPAD + t) * DD + tid] = (_Float16)Fp;
    if (tid == 0) Fn2[b * TPAD + t] = fn2;
}

// ---------------- 6) WMMA distance GEMM ----------------
__global__ void __launch_bounds__(128) k_dist(const _Float16* __restrict__ If,
                                              const float* __restrict__ In2,
                                              const _Float16* __restrict__ A,
                                              const float* __restrict__ Fn2w,
                                              float* __restrict__ out) {
    int b    = blockIdx.y;
    int tid  = threadIdx.x;
    int wave = tid >> 5, lane = tid & 31;
    int n0   = blockIdx.x * 64 + wave * 16;

    __shared__ float fn2sh[TPAD];
    if (tid < TPAD) fn2sh[tid] = Fn2w[b * TPAD + tid];
    __syncthreads();

    int lidx = lane & 15;
    int half = lane >> 4;
    int col  = n0 + lidx;
    int colC = col < NROW ? col : NROW - 1;

    const _Float16* Ic = If + (size_t)colC * DD;
    const _Float16* Ab = A + (size_t)b * TPAD * DD;
    v8f acc[4];
    acc[0] = (v8f){}; acc[1] = (v8f){}; acc[2] = (v8f){}; acc[3] = (v8f){};

#pragma unroll
    for (int ks = 0; ks < 4; ++ks) {
        const int k0 = ks * 32;
        // B fragment: lane = column, contiguous K = k0 + half*16 .. +15
        v16h bf;
        {
            const float4* p = (const float4*)(Ic + k0 + half * 16);
            float4* d = (float4*)&bf;
            d[0] = p[0]; d[1] = p[1];
        }
#pragma unroll
        for (int m = 0; m < 4; ++m) {
            // A fragment: lane = row (lidx), contiguous K = k0 + half*16 .. +15
            v16h af;
            const _Float16* ar = Ab + (size_t)(m * 16 + lidx) * DD + k0 + half * 16;
            float4* d = (float4*)&af;
            d[0] = *(const float4*)(ar);
            d[1] = *(const float4*)(ar + 8);
            acc[m] = __builtin_amdgcn_wmma_f32_16x16x32_f16(
                false, af, false, bf, (short)0, acc[m], false, false);
        }
    }

    // vI = v_b . I_col came out of the tensor core: A row 50 = v_b lives in
    // tile m=3, tile-row 2 -> acc[3][2], lanes 0..15 (cols n0..n0+15).
    // Broadcast to both lane halves with ds_bpermute (byte index, wave32).
    float vI = __int_as_float(
        __builtin_amdgcn_ds_bpermute(lidx * 4, __float_as_int(acc[3][2])));

    float vI2  = vI * vI;
    float In2c = In2[colC];
#pragma unroll
    for (int m = 0; m < 4; ++m) {
#pragma unroll
        for (int r = 0; r < 8; ++r) {
            int row = m * 16 + r + half * 8;
            if (row < TT && col < NROW) {
                float dmm = fn2sh[row] + In2c - vI2 - 2.f * acc[m][r];
                out[((size_t)(b * TT + row)) * NROW + col] = dmm;
            }
        }
    }
}

extern "C" void kernel_launch(void* const* d_in, const int* in_sizes, int n_in,
                              void* d_out, int out_size, void* d_ws, size_t ws_size,
                              hipStream_t stream) {
    (void)in_sizes; (void)n_in; (void)out_size; (void)ws_size;
    const int*   session = (const int*)d_in[0];
    const int*   lengths = (const int*)d_in[1];
    const int*   tau     = (const int*)d_in[2];
    const float* I_emb   = (const float*)d_in[5];
    const float* P_emb   = (const float*)d_in[6];
    const float* EP_emb  = (const float*)d_in[7];
    const float* ES_emb  = (const float*)d_in[8];
    const float* V       = (const float*)d_in[9];
    const float* WP1     = (const float*)d_in[10];
    const float* WP2     = (const float*)d_in[11];
    const float* Wq      = (const float*)d_in[12];
    const float* Wk      = (const float*)d_in[13];
    const float* Wv      = (const float*)d_in[14];
    const float* Wo      = (const float*)d_in[15];
    const float* FF1     = (const float*)d_in[16];
    const float* FF2     = (const float*)d_in[17];
    const float* ln1_g   = (const float*)d_in[18];
    const float* ln1_b   = (const float*)d_in[19];
    const float* ln2_g   = (const float*)d_in[20];
    const float* ln2_b   = (const float*)d_in[21];

    char* w = (char*)d_ws;
    _Float16* If  = (_Float16*)(w + OFF_IF);
    float*    In2 = (float*)(w + OFF_IN2);
    float*    Xs  = (float*)(w + OFF_XS);
    float*    Q   = (float*)(w + OFF_Q);
    float*    Kb  = (float*)(w + OFF_K);
    float*    Vb  = (float*)(w + OFF_V);
    float*    H   = (float*)(w + OFF_H);
    float*    PS  = (float*)(w + OFF_PS);
    float*    VV  = (float*)(w + OFF_VV);
    _Float16* A   = (_Float16*)(w + OFF_A);
    float*    Fn2 = (float*)(w + OFF_FN2);

    float* dist = (float*)d_out;
    float* orth = dist + DIST_ELEMS;

    k_prep <<<NROW,      128, 0, stream>>>(I_emb, If, In2);
    k_proxy<<<BB,        128, 0, stream>>>(session, lengths, tau, I_emb, P_emb, EP_emb, V,
                                           WP1, WP2, PS, VV, orth);
    k_qkv  <<<BB * TT,   128, 0, stream>>>(session, lengths, I_emb, ES_emb, Wq, Wk, Wv,
                                           Xs, Q, Kb, Vb);
    k_attn <<<BB * TT,   128, 0, stream>>>(session, Xs, Q, Kb, Vb, Wo, ln1_g, ln1_b, H);
    k_ffn  <<<BB * TPAD, 128, 0, stream>>>(session, H, FF1, FF2, ln2_g, ln2_b, PS, VV, A, Fn2);

    dim3 grid((NROW + 63) / 64, BB);
    k_dist <<<grid,      128, 0, stream>>>(If, In2, A, Fn2, dist);
}
